// BahdanauAttentionOptimized_15633680957692
// MI455X (gfx1250) — matrix operand order repaired and checked by reading, not compiled, not run on hardware
//
#include <hip/hip_runtime.h>
#include <hip/hip_bf16.h>
#include <math.h>

// ---------------------------------------------------------------------------
// Bahdanau attention, MI455X (gfx1250, wave32, WMMA + async global->LDS DMA).
// B=32, S=2048, ENC=DEC=1024.
//   out[0 .. 32767]      = context  (32 x 1024)
//   out[32768 .. 98303]  = attn_w   (32 x 2048)
// Workspace: dec_proj (32x1024 f32) + scores (32x2048 f32) + Wpk (2 MB bf16
// pre-packed B-fragment image of W_enc) ≈ 2.4 MB.
// ---------------------------------------------------------------------------

typedef __attribute__((ext_vector_type(16))) __bf16 v16bf;
typedef __attribute__((ext_vector_type(8)))  float  v8f;

#define BB   32
#define SS   2048
#define DD   1024

// LDS W-tile row stride in u32 words: 20 -> 80 B rows. (20n mod 64) covers all
// residues with spacing 4 => conflict-free ds_load_b128, and rows stay 16B
// aligned (80 ≡ 0 mod 16).
#define WROW 20

__device__ __forceinline__ __bf16 f2bf(float f) { return (__bf16)f; }
__device__ __forceinline__ unsigned short bf_bits(float f) {
  return __builtin_bit_cast(unsigned short, f2bf(f));
}

struct bqpair { uint4 a, b; };   // 32 B: bit-identical to a v16bf B fragment

// ---------------------------------------------------------------------------
// Kernel 0: pre-pack W_enc (fp32, (1024,1024) row-major) into the exact bf16
// B-fragment LDS image, tiled as [tile = nc*32+kc][chunk c = (n<<2)|q][4 u32].
// Tile = (K slab kc*32..+31) x (N slab nc*256..+255); u32 packs K-pair
// (2t, 2t+1) for column n:  word(t,n) = bf16(W[k0+2t][n]) | bf16(W[k0+2t+1][n])<<16
// Chunk c holds words t = q*4 .. q*4+3 of column n. 16 KB per tile, 2 MB total.
// ---------------------------------------------------------------------------
__global__ void __launch_bounds__(256)
k_pack_w(const float* __restrict__ Wenc, unsigned* __restrict__ Wpk)
{
  const int w     = blockIdx.x * 256 + threadIdx.x;  // 0 .. 524287
  const int tile  = w >> 12;                         // nc*32 + kc
  const int inner = w & 4095;                        // u32 within tile
  const int c     = inner >> 2;                      // chunk (n<<2 | q)
  const int i     = inner & 3;                       // word within chunk
  const int nl    = c >> 2;
  const int q     = c & 3;
  const int t     = q * 4 + i;                       // K-pair within tile
  const int nc    = tile >> 5;
  const int kc    = tile & 31;
  const int n     = nc * 256 + nl;
  const int K     = kc * 32 + 2 * t;
  const unsigned short u0 = bf_bits(Wenc[(size_t)K * DD + n]);
  const unsigned short u1 = bf_bits(Wenc[(size_t)(K + 1) * DD + n]);
  Wpk[w] = (unsigned)u0 | ((unsigned)u1 << 16);
}

// ---------------------------------------------------------------------------
// Kernel 1: dec_proj[b,d] = sum_k dec_hidden[b,k] * W_dec[k,d] + bias[d]
// ---------------------------------------------------------------------------
__global__ void __launch_bounds__(256)
k_dec_proj(const float* __restrict__ dh, const float* __restrict__ W,
           const float* __restrict__ bias, float* __restrict__ dp)
{
  const int idx = blockIdx.x * 256 + threadIdx.x;   // 0 .. 32767
  const int b = idx >> 10;
  const int d = idx & (DD - 1);
  float acc = bias[d];
  const float* h = dh + (size_t)b * DD;
#pragma unroll 4
  for (int k = 0; k < DD; ++k)
    acc = fmaf(h[k], W[(size_t)k * DD + d], acc);
  dp[idx] = acc;
}

// ---------------------------------------------------------------------------
// Async DMA: stage one 16 KB pre-packed W tile into an LDS buffer.
// 1024 chunks of 16 B; thread tid issues chunks c = j*256+tid (consecutive
// lanes -> consecutive global 16 B = perfectly coalesced), scattering to the
// padded LDS rows (dst byte = 80*n + 16*q).
// ---------------------------------------------------------------------------
__device__ __forceinline__ void issue_w_tile(const unsigned* __restrict__ Wpk,
                                             int tile, unsigned lds_base_byte,
                                             int tid)
{
  const unsigned* src = Wpk + ((size_t)tile << 12);   // 4096 u32 per tile
#pragma unroll
  for (int j = 0; j < 4; ++j) {
    const int c = j * 256 + tid;
    const unsigned dst  = lds_base_byte + 80u * (unsigned)(c >> 2)
                                        + 16u * (unsigned)(c & 3);
    const unsigned voff = (unsigned)c * 16u;
    asm volatile("global_load_async_to_lds_b128 %0, %1, %2"
                 :: "v"(dst), "v"(voff), "s"(src)
                 : "memory");
  }
}

// ---------------------------------------------------------------------------
// Kernel 2: fused  enc_proj = enc @ W_enc ;  scores = tanh(dec_proj+enc_proj)·v
// Grid: 512 blocks (32 batches x 16 s-tiles of 128 rows), 256 threads (8 waves).
// Each wave: 16 s-rows (A tile), N chunk = 256 (16 WMMA f32 accumulators),
// K chunk = 32. W tiles arrive via double-buffered async DMA from Wpk.
// B fragments are software-pipelined (2 register sets) so each WMMA overlaps
// the next tile's ds_load_b128 pair instead of stalling on s_wait_dscnt 0.
// ---------------------------------------------------------------------------
__global__ void __launch_bounds__(256)
k_scores(const float* __restrict__ enc,      // (32,2048,1024)
         const unsigned* __restrict__ Wpk,   // pre-packed bf16 W image
         const float* __restrict__ dec_proj, // (32,1024)
         const float* __restrict__ vW,       // (1024)
         float* __restrict__ scores)         // (32,2048)
{
  __shared__ __align__(16) unsigned ldsW[2][256][WROW];   // 2 x 20 KB

  const int tid  = threadIdx.x;
  const int wave = tid >> 5;
  const int lane = tid & 31;
  const int nloc = lane & 15;   // N column within tile / A row
  const int g    = lane >> 4;   // half-wave group

  const int b  = blockIdx.x >> 4;
  const int s0 = (blockIdx.x & 15) * 128;
  const int arow = s0 + wave * 16 + nloc;          // A-matrix row this lane feeds
  const float* arow_base = enc + ((size_t)b * SS + arow) * DD;

  const unsigned lds0 = (unsigned)(size_t)(void*)&ldsW[0][0][0];
  const unsigned lds1 = lds0 + 256u * WROW * 4u;

  float sacc[8];
#pragma unroll
  for (int r = 0; r < 8; ++r) sacc[r] = 0.0f;

  const v8f vzero = {0.f, 0.f, 0.f, 0.f, 0.f, 0.f, 0.f, 0.f};

  for (int nc = 0; nc < 4; ++nc) {
    v8f acc[16];
#pragma unroll
    for (int nt = 0; nt < 16; ++nt) acc[nt] = vzero;

    // prologue: DMA first K tile of this N chunk into buffer 0
    issue_w_tile(Wpk, nc * 32, lds0, tid);

    for (int kc = 0; kc < 32; ++kc) {
      const int k0  = kc * 32;
      const int buf = kc & 1;

      // my 4 async chunks for tile kc have landed; barrier -> whole tile ready
      asm volatile("s_wait_asynccnt 0x0" ::: "memory");
      __syncthreads();

      // kick DMA for the next K tile into the other buffer (overlaps WMMA)
      if (kc + 1 < 32)
        issue_w_tile(Wpk, nc * 32 + kc + 1, (kc & 1) ? lds0 : lds1, tid);
      if (kc + 1 < 32)   // warm L2 for the next A slab too
        __builtin_prefetch(arow_base + k0 + 32, 0, 0);

      // ---- A fragment: 16x32 bf16, contiguous k = g*16+s (lane m=nloc, group g) ----
      v16bf afrag;
#pragma unroll
      for (int i = 0; i < 8; ++i) {
        const int kk = g * 16 + 2 * i; // g*16+2i
        const float2 f2 = *reinterpret_cast<const float2*>(arow_base + k0 + kk);
        afrag[2 * i]     = f2bf(f2.x);
        afrag[2 * i + 1] = f2bf(f2.y);
      }

      // ---- 16 WMMA tiles across the N chunk; B frags via 2x ds_load_b128,
      //      2-deep software pipeline so WMMA nt hides loads of nt+1 ----
      const uint4* bp =
          reinterpret_cast<const uint4*>(&ldsW[buf][nloc][g * 8]);
      // tile nt lives 16*WROW u32 = 80 uint4s further down
      bqpair cur, nxt;
      cur.a = bp[0];
      cur.b = bp[1];
#pragma unroll
      for (int nt = 0; nt < 16; ++nt) {
        if (nt + 1 < 16) {
          nxt.a = bp[(nt + 1) * 80];
          nxt.b = bp[(nt + 1) * 80 + 1];
        }
        const v16bf bfrag = __builtin_bit_cast(v16bf, cur);
        acc[nt] = __builtin_amdgcn_wmma_f32_16x16x32_bf16(
            false, afrag, false, bfrag, (short)0, acc[nt], false, false);
        cur = nxt;
      }
    }

    // ---- fused epilogue: tanh(enc_proj + dec_proj) * v, per-lane partials ----
    const int n0 = nc * 256;
#pragma unroll
    for (int nt = 0; nt < 16; ++nt) {
      const int n   = n0 + nt * 16 + nloc;
      const float dp = dec_proj[(size_t)b * DD + n];
      const float vv = vW[n];
#pragma unroll
      for (int r = 0; r < 8; ++r)
        sacc[r] += tanhf(acc[nt][r] + dp) * vv;
    }
  }

  // ---- one cross-lane reduction (16-lane groups), then store 16 rows ----
#pragma unroll
  for (int r = 0; r < 8; ++r) {
    float c = sacc[r];
#pragma unroll
    for (int off = 8; off; off >>= 1)
      c += __shfl_xor(c, off, 16);
    sacc[r] = c;
  }
  if (nloc == 0) {
#pragma unroll
    for (int r = 0; r < 8; ++r)
      scores[(size_t)b * SS + s0 + wave * 16 + g * 8 + r] = sacc[r];
  }
}

// ---------------------------------------------------------------------------
// Kernel 3: masked softmax over S per batch row -> attn (written to d_out).
// ---------------------------------------------------------------------------
__global__ void __launch_bounds__(256)
k_softmax(const float* __restrict__ scores, const unsigned char* __restrict__ mask,
          float* __restrict__ attn)
{
  __shared__ float red[256];
  const int b = blockIdx.x, tid = threadIdx.x;
  float loc[8];
  float m = -INFINITY;
#pragma unroll
  for (int j = 0; j < 8; ++j) {
    const int s = j * 256 + tid;
    float v = scores[(size_t)b * SS + s];
    v = mask[(size_t)b * SS + s] ? v : -10000.0f;
    loc[j] = v;
    m = fmaxf(m, v);
  }
  red[tid] = m; __syncthreads();
  for (int off = 128; off; off >>= 1) {
    if (tid < off) red[tid] = fmaxf(red[tid], red[tid + off]);
    __syncthreads();
  }
  m = red[0]; __syncthreads();

  float sum = 0.0f;
#pragma unroll
  for (int j = 0; j < 8; ++j) {
    loc[j] = __expf(loc[j] - m);
    sum += loc[j];
  }
  red[tid] = sum; __syncthreads();
  for (int off = 128; off; off >>= 1) {
    if (tid < off) red[tid] += red[tid + off];
    __syncthreads();
  }
  const float inv = 1.0f / red[0];
#pragma unroll
  for (int j = 0; j < 8; ++j)
    attn[(size_t)b * SS + j * 256 + tid] = loc[j] * inv;
}

// ---------------------------------------------------------------------------
// Kernel 4: context[b,e] = sum_s attn[b,s] * enc[b,s,e]   (coalesced pass)
// ---------------------------------------------------------------------------
__global__ void __launch_bounds__(256)
k_context(const float* __restrict__ enc, const float* __restrict__ attn,
          float* __restrict__ context)
{
  __shared__ float wsh[256];
  const int tid = threadIdx.x;
  const int b   = blockIdx.x >> 2;
  const int e   = (blockIdx.x & 3) * 256 + tid;
  float acc = 0.0f;
  for (int sb = 0; sb < 8; ++sb) {
    __syncthreads();
    wsh[tid] = attn[(size_t)b * SS + sb * 256 + tid];
    __syncthreads();
    const float* ep = enc + ((size_t)b * SS + (size_t)sb * 256) * DD + e;
#pragma unroll 8
    for (int s = 0; s < 256; ++s)
      acc = fmaf(wsh[s], ep[(size_t)s * DD], acc);
  }
  context[(size_t)b * DD + e] = acc;
}

// ---------------------------------------------------------------------------
extern "C" void kernel_launch(void* const* d_in, const int* in_sizes, int n_in,
                              void* d_out, int out_size, void* d_ws, size_t ws_size,
                              hipStream_t stream)
{
  const float*         dec_hidden = (const float*)d_in[0];
  const float*         enc        = (const float*)d_in[1];
  const unsigned char* mask       = (const unsigned char*)d_in[2];
  const float*         attn_W     = (const float*)d_in[3];
  const float*         attn_b     = (const float*)d_in[4];
  const float*         vW         = (const float*)d_in[5];

  float* out     = (float*)d_out;
  float* context = out;                 // 32*1024
  float* attn    = out + BB * DD;       // 32*2048

  float*    dec_proj = (float*)d_ws;               // 32*1024 f32
  float*    scores   = dec_proj + BB * DD;         // 32*2048 f32
  unsigned* Wpk      = (unsigned*)(scores + BB * SS); // 512K u32 (2 MB)

  // 0) pre-pack W_enc into bf16 B-fragment image (async-DMA-ready layout)
  k_pack_w<<<(512 * 1024) / 256, 256, 0, stream>>>(attn_W + (size_t)DD * DD, Wpk);

  // 1) dec_proj = dec_hidden @ W_dec + b
  k_dec_proj<<<(BB * DD) / 256, 256, 0, stream>>>(dec_hidden, attn_W, attn_b, dec_proj);

  // 2) scores: async-DMA double-buffered WMMA GEMM + fused tanh/dot epilogue
  k_scores<<<BB * (SS / 128), 256, 0, stream>>>(enc, Wpk, dec_proj, vW, scores);

  // 3) masked softmax -> attn output
  k_softmax<<<BB, 256, 0, stream>>>(scores, mask, attn);

  // 4) context = attn @ enc
  k_context<<<BB * 4, 256, 0, stream>>>(enc, attn, context);
}
